// OuterProductMean_64707977281618
// MI455X (gfx1250) — matrix operand order repaired and checked by reading, not compiled, not run on hardware
//
#include <hip/hip_runtime.h>
#include <hip/hip_bf16.h>

typedef __bf16 v16bf __attribute__((ext_vector_type(16)));
typedef float  v8f   __attribute__((ext_vector_type(8)));
typedef unsigned int v4u __attribute__((ext_vector_type(4)));
typedef int v8i __attribute__((ext_vector_type(8)));
typedef int v4i __attribute__((ext_vector_type(4)));

#define S_DIM 128
#define R_DIM 256
#define CM    256
#define H_DIM 32
#define CZ    128
#define HH    1024
#define S_PAD 136   // 128 + 8 bf16 pad (16B per 256B row) -> TDM pad_interval=5, pad_amount=3
#define G_PAD 1032  // 1024 + 8 bf16 pad

// ---------------------------------------------------------------------------
// Kernel 0a: convert/transpose weights to bf16, K-contiguous for B-fragments.
// ---------------------------------------------------------------------------
__global__ __launch_bounds__(256) void prep_kernel(
    const float* __restrict__ w1, const float* __restrict__ w2,
    const float* __restrict__ w_out,
    __bf16* __restrict__ wT1, __bf16* __restrict__ wT2, __bf16* __restrict__ wTo)
{
    int idx = blockIdx.x * 256 + threadIdx.x;
    if (idx < CM * H_DIM) {                 // 8192
        int k = idx >> 5, c = idx & 31;     // w[k][c]
        wT1[c * CM + k] = (__bf16)w1[idx];
        wT2[c * CM + k] = (__bf16)w2[idx];
    }
    if (idx < HH * CZ) {                    // 131072
        int k = idx >> 7, z = idx & 127;    // w_out[k][z]
        wTo[z * HH + k] = (__bf16)w_out[idx];
    }
}

// ---------------------------------------------------------------------------
// Kernel 0b: norm[i][j] = sum_s mask[s][i]*mask[s][j]
// ---------------------------------------------------------------------------
__global__ __launch_bounds__(256) void norm_kernel(
    const float* __restrict__ mask, float* __restrict__ norm)
{
    int idx = blockIdx.x * 256 + threadIdx.x;   // 65536
    int i = idx >> 8, j = idx & 255;
    float s = 0.f;
    #pragma unroll 4
    for (int t = 0; t < S_DIM; ++t)
        s += mask[t * R_DIM + i] * mask[t * R_DIM + j];
    norm[idx] = s;
}

// ---------------------------------------------------------------------------
// Kernel 1: LayerNorm + dual projection (WMMA), masked, bias added.
// Block = 128 threads (4 waves), 16 flat rows. aT/bT bf16 [i][c][s].
// ---------------------------------------------------------------------------
__global__ __launch_bounds__(128) void ln_proj_kernel(
    const float* __restrict__ m,    const float* __restrict__ mask,
    const float* __restrict__ ln_g, const float* __restrict__ ln_b,
    const float* __restrict__ b1,   const float* __restrict__ b2,
    const __bf16* __restrict__ wT1, const __bf16* __restrict__ wT2,
    __bf16* __restrict__ aT, __bf16* __restrict__ bT)
{
    __shared__ __bf16 mn[16][CM + 8];   // padded rows
    const int tid  = threadIdx.x;
    const int wave = tid >> 5;
    const int lane = tid & 31;
    const int half = lane >> 4;
    const int l16  = lane & 15;
    const int r0   = blockIdx.x * 16;

    for (int q = 0; q < 4; ++q) {       // LN: 4 rows per wave
        int rl = wave * 4 + q;
        const float* row = m + (size_t)(r0 + rl) * CM;
        float vals[8];
        float s = 0.f, s2 = 0.f;
        #pragma unroll
        for (int u = 0; u < 8; ++u) {
            float v = row[lane + 32 * u];
            vals[u] = v; s += v; s2 += v * v;
        }
        #pragma unroll
        for (int off = 16; off > 0; off >>= 1) {  // wave32 tree reduce
            s  += __shfl_xor(s,  off, 32);
            s2 += __shfl_xor(s2, off, 32);
        }
        float mu   = s * (1.0f / CM);
        float var  = s2 * (1.0f / CM) - mu * mu;
        float rstd = rsqrtf(var + 1e-5f);
        #pragma unroll
        for (int u = 0; u < 8; ++u) {
            int k = lane + 32 * u;
            mn[rl][k] = (__bf16)((vals[u] - mu) * rstd * ln_g[k] + ln_b[k]);
        }
    }
    __syncthreads();

    // wave -> (projection, 16-col tile). A = mn[16x256], B = wT[256x32]
    const int proj = wave >> 1;
    const int ct   = wave & 1;
    const __bf16* wT   = proj ? wT2 : wT1;
    const float*  bias = proj ? b2  : b1;
    __bf16*       dst  = proj ? bT  : aT;

    v8f acc = {};
    for (int ks = 0; ks < 8; ++ks) {
        v16bf af, bfm;
        #pragma unroll
        for (int e = 0; e < 16; ++e) {
            int ka = ks * 32 + ((e < 8) ? (half * 8 + e) : (16 + half * 8 + (e - 8)));
            af[e] = mn[l16][ka];
            int kb = ks * 32 + ((e < 8) ? (half * 8 + e) : (16 + half * 8 + (e - 8)));
            bfm[e] = wT[(ct * 16 + l16) * CM + kb];
        }
        acc = __builtin_amdgcn_wmma_f32_16x16x32_bf16(false, af, false, bfm,
                                                      (short)0, acc, false, false);
    }

    int   c    = ct * 16 + l16;
    float bval = bias[c];
    #pragma unroll
    for (int v = 0; v < 8; ++v) {
        int r  = r0 + v + 8 * half;
        int si = r >> 8;        // flat row = s*R + i
        int ii = r & 255;
        float val = (acc[v] + bval) * mask[si * R_DIM + ii];
        dst[((size_t)ii * H_DIM + c) * S_DIM + si] = (__bf16)val;
    }
}

// ---------------------------------------------------------------------------
// TDM helper: issue tensor_load_to_lds of a [rows x 256B] tile, with 16B of
// LDS padding inserted after every 256B row (pad_interval=5 -> 32x8B,
// pad_amount=3 -> 4 DWORDs), data_size=8B. 2D tensor: groups 2/3 zero.
// This toolchain's builtin is the 6-arg form: (g0, g1, g2, g3, g?, cpol).
// ---------------------------------------------------------------------------
__device__ __forceinline__ void tdm_load_tile(const void* gsrc, unsigned lds_byte_addr,
                                              unsigned rows)
{
    unsigned long long ga = (unsigned long long)(uintptr_t)gsrc;
    v4u g0;
    g0[0] = 1u;                                   // count=1, user descriptor, load
    g0[1] = lds_byte_addr;                        // lds_addr
    g0[2] = (unsigned)(ga & 0xFFFFFFFFu);         // global_addr[31:0]
    g0[3] = (unsigned)((ga >> 32) & 0x01FFFFFFu)  // global_addr[56:32]
          | 0x80000000u;                          // type=2 ("image")
    v8i g1;
    g1[0] = (3 << 16)        // data_size = 8B
          | (1 << 20)        // pad_enable
          | (5 << 22)        // pad_interval: 32 x 8B = 256B
          | (3 << 25);       // pad_amount: 4 DWORDs = 16B
    g1[1] = (32 << 16);                           // tensor_dim0 = 32 (8B units/row)
    g1[2] = (int)(rows << 16);                    // tensor_dim1 = rows
    g1[3] = (32 << 16);                           // tile_dim0 = 32
    g1[4] = (int)rows;                            // tile_dim1 = rows (tile_dim2 = 0)
    g1[5] = 32;                                   // tensor_dim0_stride = 32 (contiguous)
    g1[6] = (32 << 16);                           // tensor_dim1_stride lo (unused for 2D)
    g1[7] = 0;
    v4i z4 = {};
    v8i z8 = {};
    __builtin_amdgcn_tensor_load_to_lds(g0, g1, z4, z4, z8, 0);
}

// ---------------------------------------------------------------------------
// Kernel 2: fused outer-product-sum + 1024->128 contraction + bias + norm.
// Block = 256 threads (8 waves), tile = 16 i x 16 j, ~304 KB dynamic LDS.
// Tiles staged by the Tensor Data Mover (pad inserted in hardware).
// ---------------------------------------------------------------------------
__global__ __launch_bounds__(256) void outer_fused_kernel(
    const __bf16* __restrict__ aT, const __bf16* __restrict__ bT,
    const __bf16* __restrict__ wTo, const float* __restrict__ b_out,
    const float* __restrict__ norm, float* __restrict__ out)
{
    extern __shared__ char smem[];
    __bf16* aL = (__bf16*)smem;              // [16][32][S_PAD]
    __bf16* bL = aL + 16 * 32 * S_PAD;       // [16][32][S_PAD]
    __bf16* gL = bL + 16 * 32 * S_PAD;       // [16 pairs][G_PAD]

    const int tid  = threadIdx.x;
    const int wave = tid >> 5;
    const int lane = tid & 31;
    const int half = lane >> 4;
    const int l16  = lane & 15;

    const int jb = blockIdx.x & 15;
    const int ib = blockIdx.x >> 4;
    const int i0 = ib * 16, j0 = jb * 16;

    // ---- TDM staging: wave 0 issues two 128KB tile DMAs (512 rows x 256B) ----
    if (wave == 0) {
        tdm_load_tile(aT + (size_t)i0 * H_DIM * S_DIM,
                      (unsigned)(uintptr_t)aL, 512u);
        tdm_load_tile(bT + (size_t)j0 * H_DIM * S_DIM,
                      (unsigned)(uintptr_t)bL, 512u);
        __builtin_amdgcn_s_wait_tensorcnt(0);
    }
    __syncthreads();

    for (int ig = 0; ig < 16; ++ig) {           // one i per pair-group (16 j's)
        // ---- Phase A: G[p] = A_i^T B_j (32x32 f32, K=S=128); 64 jobs / 8 waves ----
        for (int job = wave; job < 64; job += 8) {
            int p  = job >> 2;
            int ct = (job >> 1) & 1;
            int et = job & 1;
            const __bf16* arow = aL + (ig * 32 + ct * 16 + l16) * S_PAD; // row M=c
            const __bf16* brow = bL + (p  * 32 + et * 16 + l16) * S_PAD; // col N=e, K=s
            v8f acc = {};
            for (int ks = 0; ks < 4; ++ks) {
                v16bf af, bfm;
                #pragma unroll
                for (int e = 0; e < 16; ++e) {
                    int ka = ks * 32 + ((e < 8) ? (half * 8 + e) : (16 + half * 8 + (e - 8)));
                    af[e] = arow[ka];
                    int kb = ks * 32 + ((e < 8) ? (half * 8 + e) : (16 + half * 8 + (e - 8)));
                    bfm[e] = brow[kb];
                }
                acc = __builtin_amdgcn_wmma_f32_16x16x32_bf16(false, af, false, bfm,
                                                              (short)0, acc, false, false);
            }
            __bf16* g = gL + p * G_PAD;          // vec(G) row, bf16
            #pragma unroll
            for (int v = 0; v < 8; ++v) {
                int c = ct * 16 + v + 8 * half;
                int e = et * 16 + l16;
                g[c * 32 + e] = (__bf16)acc[v];
            }
        }
        __syncthreads();

        // ---- Phase B: [16 pairs x 1024] @ [1024 x 128] -> [16 x 128] ----
        {
            int zt = wave;                       // 8 waves x 16-column z-tiles
            v8f acc = {};
            for (int ks = 0; ks < 32; ++ks) {
                v16bf af, bfm;
                #pragma unroll
                for (int e = 0; e < 16; ++e) {
                    int ka = ks * 32 + ((e < 8) ? (half * 8 + e) : (16 + half * 8 + (e - 8)));
                    af[e] = gL[l16 * G_PAD + ka];            // row M = pair
                    int kb = ks * 32 + ((e < 8) ? (half * 8 + e) : (16 + half * 8 + (e - 8)));
                    bfm[e] = wTo[(zt * 16 + l16) * HH + kb]; // col N = z
                }
                acc = __builtin_amdgcn_wmma_f32_16x16x32_bf16(false, af, false, bfm,
                                                              (short)0, acc, false, false);
            }
            int   z  = zt * 16 + l16;
            float bz = b_out[z];
            int   i  = i0 + ig;
            #pragma unroll
            for (int v = 0; v < 8; ++v) {
                int j = j0 + v + 8 * half;               // pair p = row m
                float nm = norm[i * R_DIM + j];
                out[((size_t)i * R_DIM + j) * CZ + z] = (acc[v] + bz) / (1e-3f + nm);
            }
        }
        __syncthreads();
    }
}

// ---------------------------------------------------------------------------
extern "C" void kernel_launch(void* const* d_in, const int* in_sizes, int n_in,
                              void* d_out, int out_size, void* d_ws, size_t ws_size,
                              hipStream_t stream)
{
    const float* m     = (const float*)d_in[0];
    const float* mask  = (const float*)d_in[1];
    const float* ln_g  = (const float*)d_in[2];
    const float* ln_b  = (const float*)d_in[3];
    const float* w1    = (const float*)d_in[4];
    const float* b1    = (const float*)d_in[5];
    const float* w2    = (const float*)d_in[6];
    const float* b2    = (const float*)d_in[7];
    const float* w_out = (const float*)d_in[8];
    const float* b_out = (const float*)d_in[9];
    float* out = (float*)d_out;

    char* ws = (char*)d_ws;
    __bf16* aT   = (__bf16*)ws; ws += (size_t)R_DIM * H_DIM * S_DIM * 2; // 2 MB
    __bf16* bT   = (__bf16*)ws; ws += (size_t)R_DIM * H_DIM * S_DIM * 2; // 2 MB
    __bf16* wT1  = (__bf16*)ws; ws += (size_t)H_DIM * CM * 2;            // 16 KB
    __bf16* wT2  = (__bf16*)ws; ws += (size_t)H_DIM * CM * 2;            // 16 KB
    __bf16* wTo  = (__bf16*)ws; ws += (size_t)CZ * HH * 2;               // 256 KB
    float*  nrm  = (float*)ws;  ws += (size_t)R_DIM * R_DIM * 4;         // 256 KB

    prep_kernel<<<512, 256, 0, stream>>>(w1, w2, w_out, wT1, wT2, wTo);
    norm_kernel<<<256, 256, 0, stream>>>(mask, nrm);
    ln_proj_kernel<<<(S_DIM * R_DIM) / 16, 128, 0, stream>>>(
        m, mask, ln_g, ln_b, b1, b2, wT1, wT2, aT, bT);

    size_t sm = (size_t)(2 * 16 * 32 * S_PAD + 16 * G_PAD) * sizeof(__bf16); // ~304 KB
    (void)hipFuncSetAttribute((const void*)outer_fused_kernel,
                              hipFuncAttributeMaxDynamicSharedMemorySize, (int)sm);
    outer_fused_kernel<<<(R_DIM / 16) * (R_DIM / 16), 256, sm, stream>>>(
        aT, bT, wTo, b_out, nrm, out);
}
